// FullAttention_with_pos_emb_supply_yuyi_45981919871292
// MI455X (gfx1250) — matrix operand order repaired and checked by reading, not compiled, not run on hardware
//
#include <hip/hip_runtime.h>

typedef _Float16 v16h __attribute__((ext_vector_type(16)));
typedef _Float16 v4h  __attribute__((ext_vector_type(4)));
typedef float    v8f  __attribute__((ext_vector_type(8)));
typedef float    v4f  __attribute__((ext_vector_type(4)));

#define B_      2
#define L_      2048
#define H_      16
#define E_      64
#define ROWS_WG 128   // rows per workgroup (8 waves x 16 rows)
#define NWAVES  8
#define SBLK    32    // s-columns per block iteration (== WMMA K for f16)

// ds_swizzle XOR-butterfly step (stays inside each 16-lane half for m<16)
template <int M>
__device__ __forceinline__ float swz_max_step(float x) {
    return fmaxf(x, __int_as_float(
        __builtin_amdgcn_ds_swizzle(__float_as_int(x), (M << 10) | 0x1f)));
}

// out[b,l,h,d] = softmax_causal(QK^T/sqrt(E))@V + 0.1 * init_score@V
__global__ __launch_bounds__(256)
void fullattn_bias_kernel(const float* __restrict__ q,
                          const float* __restrict__ kk,
                          const float* __restrict__ v,
                          const float* __restrict__ initsc,
                          float* __restrict__ out)
{
    __shared__ __align__(16) _Float16 Klds[SBLK * E_];          // [s][e]  4KB
    __shared__ __align__(16) _Float16 Vt[E_ * SBLK];            // [d][s]  4KB (transposed)
    __shared__ __align__(16) _Float16 Pscr[NWAVES][16 * SBLK];  // per-wave P tile (row-major [m][k]), 8KB

    const int tid  = threadIdx.x;
    const int wave = tid >> 5;
    const int lane = tid & 31;
    const int half = lane >> 4;     // 0: lanes 0-15, 1: lanes 16-31
    const int ln   = lane & 15;

    const int l0 = blockIdx.x * ROWS_WG;
    const int h  = blockIdx.y;
    const int b  = blockIdx.z;
    const int lw = l0 + wave * 16;  // first row owned by this wave

    const float kScale = 0.18033688011112042f;  // (1/sqrt(64)) * log2(e)

    // ---- Q in WMMA A-layout, pre-scaled by kScale (exp2-domain scores for free) ----
    // lane<16 : m=lane,    e c+0..15
    // lane>=16: m=lane-16, e c+16..31
    v16h qa[2];
    {
        const float* qrow = q + (((size_t)b * L_ + (lw + ln)) * H_ + h) * E_;
        #pragma unroll
        for (int c = 0; c < 2; ++c) {
            const int e0 = c * 32 + half * 16;
            v4f a0 = *(const v4f*)(qrow + e0);
            v4f a1 = *(const v4f*)(qrow + e0 + 4);
            v4f a2 = *(const v4f*)(qrow + e0 + 8);
            v4f a3 = *(const v4f*)(qrow + e0 + 12);
            #pragma unroll
            for (int i = 0; i < 4; ++i) {
                qa[c][i]      = (_Float16)(a0[i] * kScale);
                qa[c][i + 4]  = (_Float16)(a1[i] * kScale);
                qa[c][i + 8]  = (_Float16)(a2[i] * kScale);
                qa[c][i + 12] = (_Float16)(a3[i] * kScale);
            }
        }
    }

    // all-ones B tile: row-sum of P via the matrix unit
    v16h onesb;
    #pragma unroll
    for (int i = 0; i < 16; ++i) onesb[i] = (_Float16)1.0f;

    // ---- accumulators ----
    v8f o_acc[4]    = {};   // attention numerator, four 16-wide d-groups
    v8f bias_acc[4] = {};   // init_score @ V
    v8f lsum_acc    = {};   // softmax denominator (P @ ones), same row mapping
    float m_run[8];
    #pragma unroll
    for (int r = 0; r < 8; ++r) m_run[r] = -1e30f;

    const int myDiag = lw + 15;

    // staging geometry: 512 float4 chunks / 256 threads = 2 per thread
    const int sst = (tid >> 4);            // s row for jj=0 (0..15), +16 for jj=1
    const int dst = (tid & 15) * 4;        // d/e column (multiple of 4)

    for (int s0 = 0; s0 < L_; s0 += SBLK) {
        const bool needK = (s0 < l0 + ROWS_WG);               // workgroup-uniform
        const size_t gblk = (((size_t)b * L_ + (size_t)s0) * H_ + h) * (size_t)E_;

        __syncthreads();
        // ---- stage V transposed (always) ----
        #pragma unroll
        for (int jj = 0; jj < 2; ++jj) {
            const int s = sst + 16 * jj;
            v4f val = *(const v4f*)(v + gblk + (size_t)s * (H_ * E_) + dst);
            Vt[(dst + 0) * SBLK + s] = (_Float16)val[0];
            Vt[(dst + 1) * SBLK + s] = (_Float16)val[1];
            Vt[(dst + 2) * SBLK + s] = (_Float16)val[2];
            Vt[(dst + 3) * SBLK + s] = (_Float16)val[3];
        }
        // ---- stage K row-major (causal range only; single uniform branch) ----
        if (needK) {
            #pragma unroll
            for (int jj = 0; jj < 2; ++jj) {
                const int s = sst + 16 * jj;
                v4f val = *(const v4f*)(kk + gblk + (size_t)s * (H_ * E_) + dst);
                v4h kp;
                kp[0] = (_Float16)val[0]; kp[1] = (_Float16)val[1];
                kp[2] = (_Float16)val[2]; kp[3] = (_Float16)val[3];
                *(v4h*)&Klds[s * E_ + dst] = kp;
            }
        }
        // ---- prefetch next block's K/V lines (global_prefetch_b8) ----
        if (s0 + SBLK < L_) {
            const size_t gnext = gblk + (size_t)SBLK * (H_ * E_);
            __builtin_prefetch(v + gnext + (size_t)sst * (H_ * E_) + dst, 0, 1);
            __builtin_prefetch(v + gnext + (size_t)(sst + 16) * (H_ * E_) + dst, 0, 1);
            if (s0 + SBLK < l0 + ROWS_WG) {
                __builtin_prefetch(kk + gnext + (size_t)sst * (H_ * E_) + dst, 0, 1);
                __builtin_prefetch(kk + gnext + (size_t)(sst + 16) * (H_ * E_) + dst, 0, 1);
            }
        }
        __syncthreads();

        // ---- init_score tile in A-layout (vectorized) ----
        v16h ia;
        {
            const float* irow = initsc + ((size_t)b * L_ + (lw + ln)) * L_ + s0 + half * 16;
            v4f a0 = *(const v4f*)(irow);
            v4f a1 = *(const v4f*)(irow + 4);
            v4f a2 = *(const v4f*)(irow + 8);
            v4f a3 = *(const v4f*)(irow + 12);
            #pragma unroll
            for (int i = 0; i < 4; ++i) {
                ia[i]      = (_Float16)a0[i];
                ia[i + 4]  = (_Float16)a1[i];
                ia[i + 8]  = (_Float16)a2[i];
                ia[i + 12] = (_Float16)a3[i];
            }
        }

        const bool doAttn = (s0 <= myDiag);   // wave-uniform

        if (doAttn) {
            // ---- scores (already in exp2 domain thanks to pre-scaled Q) ----
            v8f s_t[2];
            #pragma unroll
            for (int g2 = 0; g2 < 2; ++g2) {
                v8f acc = {};
                #pragma unroll
                for (int c = 0; c < 2; ++c) {
                    v16h kb = *(const v16h*)&Klds[(g2 * 16 + ln) * E_ + c * 32 + half * 16];
                    acc = __builtin_amdgcn_wmma_f32_16x16x32_f16(
                        false, qa[c], false, kb, (short)0, acc, false, false);
                }
                s_t[g2] = acc;
            }

            // ---- online softmax; reg r <-> row r+8*half ----
            float alpha[8];
            const bool edge = (s0 + SBLK - 1 > lw);   // wave-uniform: mask needed?
            if (edge) {
                #pragma unroll
                for (int r = 0; r < 8; ++r) {
                    const int lrow = lw + r + 8 * half;
                    const float f0 = ((s0 + ln)      <= lrow) ? 1.0f : 0.0f;
                    const float f1 = ((s0 + 16 + ln) <= lrow) ? 1.0f : 0.0f;
                    float x0 = (f0 != 0.0f) ? s_t[0][r] : -1e30f;
                    float x1 = (f1 != 0.0f) ? s_t[1][r] : -1e30f;
                    float bm = fmaxf(x0, x1);
                    bm = swz_max_step<1>(bm);
                    bm = swz_max_step<2>(bm);
                    bm = swz_max_step<4>(bm);
                    bm = swz_max_step<8>(bm);
                    const float mn = fmaxf(m_run[r], bm);
                    alpha[r]  = exp2f(m_run[r] - mn);
                    m_run[r]  = mn;
                    s_t[0][r] = exp2f(x0 - mn) * f0;   // branchless
                    s_t[1][r] = exp2f(x1 - mn) * f1;
                }
            } else {
                #pragma unroll
                for (int r = 0; r < 8; ++r) {
                    float bm = fmaxf(s_t[0][r], s_t[1][r]);
                    bm = swz_max_step<1>(bm);
                    bm = swz_max_step<2>(bm);
                    bm = swz_max_step<4>(bm);
                    bm = swz_max_step<8>(bm);
                    const float mn = fmaxf(m_run[r], bm);
                    alpha[r]  = exp2f(m_run[r] - mn);
                    m_run[r]  = mn;
                    s_t[0][r] = exp2f(s_t[0][r] - mn);
                    s_t[1][r] = exp2f(s_t[1][r] - mn);
                }
            }

            // P -> per-wave LDS scratch, row-major [m][k] so the A-layout
            // reload is one contiguous 32B read:
            //   perm(k): k (identity)
            _Float16* pw = &Pscr[wave][0];
            const int off0 = ln;   // perm(ln)
            #pragma unroll
            for (int r = 0; r < 8; ++r) {
                const int m = r + 8 * half;
                pw[m * SBLK + off0]     = (_Float16)s_t[0][r];
                pw[m * SBLK + off0 + 16] = (_Float16)s_t[1][r];
            }

            // rescale running numerator and denominator
            #pragma unroll
            for (int g = 0; g < 4; ++g)
                #pragma unroll
                for (int r = 0; r < 8; ++r)
                    o_acc[g][r] *= alpha[r];
            #pragma unroll
            for (int r = 0; r < 8; ++r)
                lsum_acc[r] *= alpha[r];

            // reload P in A-layout: single contiguous v16h
            const v16h pa = *(const v16h*)(pw + ln * SBLK + half * 16);

            // denominator via matrix unit: lsum += P @ ones
            lsum_acc = __builtin_amdgcn_wmma_f32_16x16x32_f16(
                false, pa, false, onesb, (short)0, lsum_acc, false, false);

            // shared V B-operand feeds both GEMMs
            #pragma unroll
            for (int g = 0; g < 4; ++g) {
                v16h vb = *(const v16h*)&Vt[(g * 16 + ln) * SBLK + half * 16];
                bias_acc[g] = __builtin_amdgcn_wmma_f32_16x16x32_f16(
                    false, ia, false, vb, (short)0, bias_acc[g], false, false);
                o_acc[g] = __builtin_amdgcn_wmma_f32_16x16x32_f16(
                    false, pa, false, vb, (short)0, o_acc[g], false, false);
            }
        } else {
            // bias-only range
            #pragma unroll
            for (int g = 0; g < 4; ++g) {
                v16h vb = *(const v16h*)&Vt[(g * 16 + ln) * SBLK + half * 16];
                bias_acc[g] = __builtin_amdgcn_wmma_f32_16x16x32_f16(
                    false, ia, false, vb, (short)0, bias_acc[g], false, false);
            }
        }
    }

    // ---- epilogue: out = O/l + 0.1*bias ----
    #pragma unroll
    for (int r = 0; r < 8; ++r) {
        const int m = r + 8 * half;
        const float invl = 1.0f / lsum_acc[r];
        float* orow = out + (((size_t)b * L_ + (lw + m)) * H_ + h) * E_;
        #pragma unroll
        for (int g = 0; g < 4; ++g)
            orow[g * 16 + ln] = o_acc[g][r] * invl + 0.1f * bias_acc[g][r];
    }
}

extern "C" void kernel_launch(void* const* d_in, const int* in_sizes, int n_in,
                              void* d_out, int out_size, void* d_ws, size_t ws_size,
                              hipStream_t stream) {
    (void)in_sizes; (void)n_in; (void)out_size; (void)d_ws; (void)ws_size;
    const float* q      = (const float*)d_in[0];
    const float* k      = (const float*)d_in[1];
    const float* v      = (const float*)d_in[2];
    // d_in[3] = attn_mask (bool) unused: causal mask computed analytically
    const float* initsc = (const float*)d_in[4];
    float* out = (float*)d_out;

    dim3 grid(L_ / ROWS_WG, H_, B_);   // (16, 16, 2)
    dim3 block(256);
    fullattn_bias_kernel<<<grid, block, 0, stream>>>(q, k, v, initsc, out);
}
